// multipole_interaction_9216999817579
// MI455X (gfx1250) — matrix operand order repaired and checked by reading, not compiled, not run on hardware
//
#include <hip/hip_runtime.h>

// MI455X (gfx1250) fused multipole-interaction pipeline.
// 16x 128x128 GEMMs per row-tile done with v_wmma_f32_16x16x32_bf16 (bf16 in,
// fp32 accumulate); elementwise T0..T3 tensor contractions in fp32 VALU in
// C-fragment layout. Weights transposed+converted to bf16 in LDS once per block.
// silu uses v_exp_f32 + v_rcp_f32 (no IEEE division expansion).

#define NPTS 50000
#define FDIM 128
#define ADIM 128

typedef __bf16 bf16;
typedef __attribute__((ext_vector_type(16))) __bf16 v16bf;
typedef __attribute__((ext_vector_type(8)))  float  v8f;

#define WSTRIDE 136                 // padded row stride (elements) for LDS tiles
#define WMAT_EL (FDIM * WSTRIDE)    // one 128x128 weight (transposed) in LDS
#define ACT_EL  (16 * WSTRIDE)      // one 16x128 activation tile in LDS
#define PW_MU_EL (3 * ACT_EL)
#define PW_T_F   (16 * 44)          // per-row T0|T1|T2|T3 packed, stride 44 floats
#define WAVES_A 4
#define WAVES_B 8

union AB16 { v16bf v; bf16 h[16]; uint4 q[2]; };

__device__ inline v8f wmma_bf16(v16bf a, v16bf b, v8f c) {
  return __builtin_amdgcn_wmma_f32_16x16x32_bf16(false, a, false, b, (short)0, c,
                                                 false, false);
}

// silu via hardware transcendentals: x * v_rcp_f32(1 + v_exp_f32(-x)).
__device__ inline float silu(float x) {
  return x * __builtin_amdgcn_rcpf(1.0f + __expf(-x));
}

// Stage W[k][n] (row-major f32) -> LDS transposed bf16 dst[n*WSTRIDE + k].
__device__ inline void stage_wT(const float* __restrict__ W, bf16* dst,
                                int tid, int nthr) {
  for (int i = tid; i < FDIM * FDIM; i += nthr) {
    int k = i >> 7, n = i & (FDIM - 1);
    dst[n * WSTRIDE + k] = (bf16)W[i];
  }
}

// B fragment: column n = ft*16 + (lane&15); K = s*32 + {0..7 | 16..23}
// (+8 for lanes 16-31); from transposed-weight LDS.
__device__ inline v16bf load_bfrag(const bf16* wt, int ft, int s, int lane) {
  int n = ft * 16 + (lane & 15);
  int kb = s * 32 + ((lane >> 4) << 3);
  const bf16* p = wt + n * WSTRIDE + kb;
  AB16 f;
  f.q[0] = *(const uint4*)(p);
  f.q[1] = *(const uint4*)(p + 16);
  return f.v;
}

// A fragment from a padded 16x128 bf16 LDS tile: row m=lane&15,
// chunks K = s*32 + {0..7 | 16..23} (+8 for lanes 16-31).
__device__ inline v16bf load_afrag_lds(const bf16* act, int s, int lane) {
  int m = lane & 15, hi = (lane >> 4) * 8;
  const bf16* p = act + m * WSTRIDE + s * 32 + hi;
  AB16 f;
  f.q[0] = *(const uint4*)(p);
  f.q[1] = *(const uint4*)(p + 16);
  return f.v;
}

// A fragment straight from global f32 row-major [.,128] with cvt to bf16.
__device__ inline v16bf load_afrag_g32(const float* base, int s, int lane) {
  int m = lane & 15, hi = (lane >> 4) * 8;
  const float* p = base + m * FDIM + s * 32 + hi;
  AB16 f;
#pragma unroll
  for (int j = 0; j < 8; ++j) f.h[j] = (bf16)p[j];
#pragma unroll
  for (int j = 0; j < 8; ++j) f.h[8 + j] = (bf16)p[16 + j];
  return f.v;
}

// A fragment straight from global bf16 row-major [.,128] (no cvt needed).
__device__ inline v16bf load_afrag_gbf(const bf16* base, int s, int lane) {
  int m = lane & 15, hi = (lane >> 4) * 8;
  const bf16* p = base + m * FDIM + s * 32 + hi;
  AB16 f;
  f.q[0] = *(const uint4*)(p);
  f.q[1] = *(const uint4*)(p + 16);
  return f.v;
}

// ---------------- Kernel A: M0..M3 + tensor contractions -> ms, mv (bf16) ----
__global__ void __launch_bounds__(WAVES_A * 32) kernA(
    const float* __restrict__ feat, const float* __restrict__ mu,
    const float* __restrict__ T0, const float* __restrict__ T1,
    const float* __restrict__ T2, const float* __restrict__ T3,
    const float* __restrict__ Wq1, const float* __restrict__ bq1,
    const float* __restrict__ Wq2, const float* __restrict__ bq2,
    const float* __restrict__ WT1, const float* __restrict__ WT2,
    const float* __restrict__ WT3,
    bf16* __restrict__ msG, bf16* __restrict__ mvG, int ntiles) {
  extern __shared__ char smem[];
  bf16* wq1 = (bf16*)smem;
  bf16* wq2 = wq1 + WMAT_EL;
  bf16* wt1 = wq1 + 2 * WMAT_EL;
  bf16* wt2 = wq1 + 3 * WMAT_EL;
  bf16* wt3 = wq1 + 4 * WMAT_EL;
  int wave = threadIdx.x >> 5, lane = threadIdx.x & 31;
  char* pw = smem + (size_t)5 * WMAT_EL * 2 +
             (size_t)wave * (PW_MU_EL * 2 + ACT_EL * 2 + PW_T_F * 4);
  bf16*  muB  = (bf16*)pw;           // 3 channel tiles
  bf16*  actH = muB + PW_MU_EL;      // H = silu(feat@Wq1+b)
  float* tB   = (float*)(actH + ACT_EL);

  int tid = threadIdx.x, nthr = blockDim.x;
  stage_wT(Wq1, wq1, tid, nthr);
  stage_wT(Wq2, wq2, tid, nthr);
  stage_wT(WT1, wt1, tid, nthr);
  stage_wT(WT2, wt2, tid, nthr);
  stage_wT(WT3, wt3, tid, nthr);
  __syncthreads();

  int tile = blockIdx.x * WAVES_A + wave;
  if (tile >= ntiles) return;
  int n0 = tile * 16;
  int col = lane & 15, hi8 = (lane >> 4) * 8;

  // Stage mu tile, channel-separated bf16.
  const float* muT = mu + (size_t)n0 * (FDIM * 3);
  for (int i = lane; i < 16 * FDIM * 3; i += 32) {
    int m = i / (FDIM * 3), r = i - m * (FDIM * 3);
    int f = r / 3, c = r - f * 3;
    muB[c * ACT_EL + m * WSTRIDE + f] = (bf16)muT[i];
  }
  // Stage per-row T data: [T0 | T1(3) | T2(9) | T3(27)] at stride 44.
  for (int i = lane; i < 16 * 40; i += 32) {
    int m = i / 40, j = i - m * 40, n = n0 + m;
    float v;
    if (j == 0)      v = T0[n];
    else if (j < 4)  v = T1[n * 3 + (j - 1)];
    else if (j < 13) v = T2[n * 9 + (j - 4)];
    else             v = T3[n * 27 + (j - 13)];
    tB[m * 44 + j] = v;
  }

  // feat A-fragments (global f32 -> bf16 in regs).
  v16bf fa[4];
  const float* fR = feat + (size_t)n0 * FDIM;
#pragma unroll
  for (int s = 0; s < 4; ++s) fa[s] = load_afrag_g32(fR, s, lane);

  // GEMM1: H = silu(feat @ Wq1 + bq1) -> actH (bf16).
  for (int ft = 0; ft < 8; ++ft) {
    v8f acc = {};
#pragma unroll
    for (int s = 0; s < 4; ++s)
      acc = wmma_bf16(fa[s], load_bfrag(wq1, ft, s, lane), acc);
    int g = ft * 16 + col;
    float b = bq1[g];
#pragma unroll
    for (int r = 0; r < 8; ++r)
      actH[(r + hi8) * WSTRIDE + g] = (bf16)silu(acc[r] + b);
  }

  const size_t plane = (size_t)NPTS * FDIM;
  for (int ft = 0; ft < 8; ++ft) {
    int g = ft * 16 + col;
    // M0 = H @ Wq2 (+bq2 later)
    v8f M0 = {};
#pragma unroll
    for (int s = 0; s < 4; ++s)
      M0 = wmma_bf16(load_afrag_lds(actH, s, lane),
                     load_bfrag(wq2, ft, s, lane), M0);
    float b2 = bq2[g];
    // M1/M2/M3 per coordinate channel
    v8f M1[3], M2[3], M3[3];
#pragma unroll
    for (int c = 0; c < 3; ++c) {
      v8f a1 = {}, a2 = {}, a3 = {};
#pragma unroll
      for (int s = 0; s < 4; ++s) {
        v16bf am = load_afrag_lds(muB + c * ACT_EL, s, lane);
        a1 = wmma_bf16(am, load_bfrag(wt1, ft, s, lane), a1);
        a2 = wmma_bf16(am, load_bfrag(wt2, ft, s, lane), a2);
        a3 = wmma_bf16(am, load_bfrag(wt3, ft, s, lane), a3);
      }
      M1[c] = a1; M2[c] = a2; M3[c] = a3;
    }
    // Elementwise tensor contractions in fp32, C-fragment layout.
#pragma unroll
    for (int r = 0; r < 8; ++r) {
      int m = r + hi8;
      const float* t = tB + m * 44;
      float t0 = t[0];
      float t1v[3], t2v[9], t3v[27];
#pragma unroll
      for (int i = 0; i < 3; ++i)  t1v[i] = t[1 + i];
#pragma unroll
      for (int i = 0; i < 9; ++i)  t2v[i] = t[4 + i];
#pragma unroll
      for (int i = 0; i < 27; ++i) t3v[i] = t[13 + i];

      float m0v = M0[r] + b2;
      float e1[3] = {M1[0][r], M1[1][r], M1[2][r]};
      float e2[3] = {M2[0][r], M2[1][r], M2[2][r]};
      float e3[3] = {M3[0][r], M3[1][r], M3[2][r]};

      float msv = t0 * m0v;
#pragma unroll
      for (int c = 0; c < 3; ++c) msv += e1[c] * t1v[c];
#pragma unroll
      for (int i = 0; i < 3; ++i) {           // M2.M2 : T2
        float qi = 0.f;
#pragma unroll
        for (int j = 0; j < 3; ++j) qi += t2v[i * 3 + j] * e2[j];
        msv += e2[i] * qi;
      }
#pragma unroll
      for (int i = 0; i < 3; ++i) {           // M3.M3.M3 : T3
        float vi = 0.f;
#pragma unroll
        for (int j = 0; j < 3; ++j) {
          float wij = 0.f;
#pragma unroll
          for (int k = 0; k < 3; ++k) wij += t3v[(i * 3 + j) * 3 + k] * e3[k];
          vi += e3[j] * wij;
        }
        msv += e3[i] * vi;
      }
      float mvv[3];
#pragma unroll
      for (int c = 0; c < 3; ++c) {
        float acc = t1v[c] * m0v;
#pragma unroll
        for (int j = 0; j < 3; ++j) acc += t2v[c * 3 + j] * e1[j];
#pragma unroll
        for (int j = 0; j < 3; ++j) {
          float wj = 0.f;
#pragma unroll
          for (int k = 0; k < 3; ++k) wj += t3v[(c * 3 + j) * 3 + k] * e2[k];
          acc += e2[j] * wj;
        }
        mvv[c] = acc;
      }
      size_t o = (size_t)(n0 + m) * FDIM + g;
      msG[o]             = (bf16)msv;
      mvG[o]             = (bf16)mvv[0];
      mvG[plane + o]     = (bf16)mvv[1];
      mvG[2 * plane + o] = (bf16)mvv[2];
    }
  }
}

// ---------------- Kernel B: ms_out / mv_out projections -----------------------
__global__ void __launch_bounds__(WAVES_B * 32) kernB(
    const bf16* __restrict__ msG, const bf16* __restrict__ mvG,
    const float* __restrict__ Ws1, const float* __restrict__ bs1,
    const float* __restrict__ Ws2, const float* __restrict__ bs2,
    const float* __restrict__ Wv, float* __restrict__ out, int ntiles) {
  extern __shared__ char smem[];
  bf16* ws1 = (bf16*)smem;
  bf16* ws2 = ws1 + WMAT_EL;
  bf16* wv  = ws1 + 2 * WMAT_EL;
  int wave = threadIdx.x >> 5, lane = threadIdx.x & 31;
  bf16* actP = (bf16*)(smem + (size_t)3 * WMAT_EL * 2) + (size_t)wave * ACT_EL;

  int tid = threadIdx.x, nthr = blockDim.x;
  stage_wT(Ws1, ws1, tid, nthr);
  stage_wT(Ws2, ws2, tid, nthr);
  stage_wT(Wv,  wv,  tid, nthr);
  __syncthreads();

  int tile = blockIdx.x * WAVES_B + wave;
  if (tile >= ntiles) return;
  int n0 = tile * 16;
  int col = lane & 15, hi8 = (lane >> 4) * 8;

  // P = silu(ms @ Ws1 + bs1) -> actP
  v16bf af[4];
  const bf16* msR = msG + (size_t)n0 * FDIM;
#pragma unroll
  for (int s = 0; s < 4; ++s) af[s] = load_afrag_gbf(msR, s, lane);
  for (int ft = 0; ft < 8; ++ft) {
    v8f acc = {};
#pragma unroll
    for (int s = 0; s < 4; ++s)
      acc = wmma_bf16(af[s], load_bfrag(ws1, ft, s, lane), acc);
    int g = ft * 16 + col;
    float b = bs1[g];
#pragma unroll
    for (int r = 0; r < 8; ++r)
      actP[(r + hi8) * WSTRIDE + g] = (bf16)silu(acc[r] + b);
  }
  // ms_out = P @ Ws2 + bs2  -> out[n][a]
  for (int ft = 0; ft < 8; ++ft) {
    v8f acc = {};
#pragma unroll
    for (int s = 0; s < 4; ++s)
      acc = wmma_bf16(load_afrag_lds(actP, s, lane),
                      load_bfrag(ws2, ft, s, lane), acc);
    int g = ft * 16 + col;
    float b = bs2[g];
#pragma unroll
    for (int r = 0; r < 8; ++r)
      out[(size_t)(n0 + r + hi8) * ADIM + g] = acc[r] + b;
  }
  // mv_out[n][a][c] = (mv_c @ Wv)
  float* out2 = out + (size_t)NPTS * ADIM;
  const size_t plane = (size_t)NPTS * FDIM;
#pragma unroll
  for (int c = 0; c < 3; ++c) {
    const bf16* mvR = mvG + c * plane + (size_t)n0 * FDIM;
#pragma unroll
    for (int s = 0; s < 4; ++s) af[s] = load_afrag_gbf(mvR, s, lane);
    for (int ft = 0; ft < 8; ++ft) {
      v8f acc = {};
#pragma unroll
      for (int s = 0; s < 4; ++s)
        acc = wmma_bf16(af[s], load_bfrag(wv, ft, s, lane), acc);
      int g = ft * 16 + col;
#pragma unroll
      for (int r = 0; r < 8; ++r)
        out2[(size_t)(n0 + r + hi8) * (ADIM * 3) + g * 3 + c] = acc[r];
    }
  }
}

extern "C" void kernel_launch(void* const* d_in, const int* in_sizes, int n_in,
                              void* d_out, int out_size, void* d_ws,
                              size_t ws_size, hipStream_t stream) {
  const float* feat = (const float*)d_in[0];
  const float* mu   = (const float*)d_in[1];
  const float* T0   = (const float*)d_in[2];
  const float* T1   = (const float*)d_in[3];
  const float* T2   = (const float*)d_in[4];
  const float* T3   = (const float*)d_in[5];
  const float* Wq1  = (const float*)d_in[6];
  const float* bq1  = (const float*)d_in[7];
  const float* Wq2  = (const float*)d_in[8];
  const float* bq2  = (const float*)d_in[9];
  const float* WT1  = (const float*)d_in[10];
  const float* WT2  = (const float*)d_in[11];
  const float* WT3  = (const float*)d_in[12];
  const float* Ws1  = (const float*)d_in[13];
  const float* bs1  = (const float*)d_in[14];
  const float* Ws2  = (const float*)d_in[15];
  const float* bs2  = (const float*)d_in[16];
  const float* Wv   = (const float*)d_in[17];
  float* out = (float*)d_out;

  // Workspace: ms (bf16 [N,128]) + mv (bf16 [3][N,128]) = 51.2 MB.
  bf16* msG = (bf16*)d_ws;
  bf16* mvG = msG + (size_t)NPTS * FDIM;

  int ntiles = NPTS / 16;  // 3125, exact
  size_t ldsA = (size_t)5 * WMAT_EL * 2 +
                (size_t)WAVES_A * (PW_MU_EL * 2 + ACT_EL * 2 + PW_T_F * 4);
  size_t ldsB = (size_t)3 * WMAT_EL * 2 + (size_t)WAVES_B * ACT_EL * 2;
  (void)hipFuncSetAttribute(reinterpret_cast<const void*>(kernA),
                            hipFuncAttributeMaxDynamicSharedMemorySize,
                            (int)ldsA);
  (void)hipFuncSetAttribute(reinterpret_cast<const void*>(kernB),
                            hipFuncAttributeMaxDynamicSharedMemorySize,
                            (int)ldsB);
  int blkA = (ntiles + WAVES_A - 1) / WAVES_A;  // 782
  int blkB = (ntiles + WAVES_B - 1) / WAVES_B;  // 391

  kernA<<<blkA, WAVES_A * 32, ldsA, stream>>>(feat, mu, T0, T1, T2, T3, Wq1,
                                              bq1, Wq2, bq2, WT1, WT2, WT3,
                                              msG, mvG, ntiles);
  kernB<<<blkB, WAVES_B * 32, ldsB, stream>>>(msG, mvG, Ws1, bs1, Ws2, bs2, Wv,
                                              out, ntiles);
}
